// LSTMPlant_88536455840160
// MI455X (gfx1250) — matrix operand order repaired and checked by reading, not compiled, not run on hardware
//
#include <hip/hip_runtime.h>
#include <hip/hip_bf16.h>

// ---------------------------------------------------------------------------
// Seq2seq LSTM for MI455X (gfx1250), bf16 WMMA with fp32 accumulation.
//   B=256, TIN=512, TT=256, H=256, gates=4H=1024
// Per-step GEMM: g[256 x 1024] = [x | h] @ Wcat^T + b   (K=352 or 512)
// 256 waves/step: wave (mt,jt) owns batch rows mt*16..+16, hidden cols
// jt*16..+16 for ALL FOUR gates, fuses the LSTM cell update in the epilogue.
// v3: distance-2 register pipeline with __builtin_amdgcn_sched_barrier(0)
//     fences so the scheduler cannot re-serialize load->wait->wmma; A tile
//     staged in LDS (ds_load path), B streams from L2-resident weights.
// ---------------------------------------------------------------------------

#define B_   256
#define TIN_ 512
#define TT_  256
#define H_   256
#define G4H_ 1024
#define NPV_ 8

typedef __attribute__((ext_vector_type(16))) __bf16 v16bf;
typedef __attribute__((ext_vector_type(8)))  float  v8f;
typedef unsigned short u16;

union BFrag { uint4 u[2]; v16bf v; };

__device__ __forceinline__ u16 f2bf(float x) {
  unsigned u = __float_as_uint(x);
  u += 0x7FFFu + ((u >> 16) & 1u);   // round-to-nearest-even
  return (u16)(u >> 16);
}
__device__ __forceinline__ float sigmoidf_(float x) { return 1.0f / (1.0f + __expf(-x)); }

// A-matrix 16x32 bf16 fragment (ISA 7.12.2): lanes 0-15 hold K 0-7 & 16-23,
// lanes 16-31 hold K 8-15 & 24-31, for row M = lane%16.
__device__ __forceinline__ v16bf loadA(const u16* p, int khalf) {
  BFrag f;
  f.u[0] = *(const uint4*)(p + khalf);        // K [khalf, khalf+8)
  f.u[1] = *(const uint4*)(p + 16 + khalf);   // K [16+khalf, 16+khalf+8)
  return f.v;
}
// B-matrix 32x16 bf16 fragment: lane holds column N = lane%16; lanes 0-15
// carry K 0-7 & 16-23, lanes 16-31 carry K 8-15 & 24-31 (same as A).
__device__ __forceinline__ v16bf loadB(const u16* p, int ksel) {
  BFrag f;
  const u16* q = p + (ksel >> 1);
  f.u[0] = *(const uint4*)(q);
  f.u[1] = *(const uint4*)(q + 16);
  return f.v;
}

// One fused LSTM layer step. Activation rows come from two K-concatenated
// sources: cols [0,LEN0) from a0 (e.g. staged x_t), cols [LEN0,K) from a1
// (recurrent h, ping-pong buffer). W is [1024][K] bf16 row-major (K padded to
// a multiple of 32 with zeros, matching zero-padded activation columns).
template<int K, int LEN0>
__global__ __launch_bounds__(256) void lstm_step(
    const u16* __restrict__ a0, int sa0,
    const u16* __restrict__ a1, int sa1,
    const u16* __restrict__ W,
    const float* __restrict__ bias,     // fused b_ih+b_hh, [1024]
    float* __restrict__ c,              // [B][H] fp32 cell state (in/out)
    u16*  __restrict__ hA, int shA,     // bf16 h out (ptr pre-offset to col)
    u16*  __restrict__ hB, int shB,     // optional second bf16 h out
    float* __restrict__ hF)             // optional fp32 h out (for FC head)
{
  // All 8 waves of a block share the same mt (8 divides 16 jt per mt).
  const int wv   = blockIdx.x * 8 + ((int)threadIdx.x >> 5);    // 0..255
  const int lane = threadIdx.x & 31;
  const int mt = wv >> 4, jt = wv & 15;
  const int l15 = lane & 15;
  const int hi  = lane >> 4;
  const int khalf = hi << 3;
  const int ksel  = hi << 4;
  const int ncol  = jt * 16 + l15;

  // ---- stage the block's 16xK A tile into LDS (b128 cooperative fill) ----
  __shared__ u16 sA[16 * K];
  {
    constexpr int N0 = LEN0 / 8;          // uint4 chunks from source 0
    constexpr int N1 = (K - LEN0) / 8;    // uint4 chunks from source 1
    uint4* d = (uint4*)sA;
    for (int i = threadIdx.x; i < 16 * N0; i += 256) {
      int r = i / N0, q = i % N0;
      d[r * (K / 8) + q] = *(const uint4*)(a0 + (size_t)(mt * 16 + r) * sa0 + q * 8);
    }
    if (N1 > 0) {
      for (int i = threadIdx.x; i < 16 * N1; i += 256) {
        int r = i / N1, q = i % N1;
        d[r * (K / 8) + N0 + q] = *(const uint4*)(a1 + (size_t)(mt * 16 + r) * sa1 + q * 8);
      }
    }
    __syncthreads();
  }
  const u16* sArow = sA + l15 * K;

  v8f acc[4];
#pragma unroll
  for (int g = 0; g < 4; ++g) {
    float bg = bias[g * H_ + ncol];
#pragma unroll
    for (int r = 0; r < 8; ++r) acc[g][r] = bg;
  }
  const u16* wrow[4];
#pragma unroll
  for (int g = 0; g < 4; ++g) wrow[g] = W + (size_t)(g * H_ + ncol) * K;

  // ---- distance-2 software pipeline, schedule pinned with sched_barrier ----
  // Iteration k issues loads for k+2, then the 4 WMMAs for k. Two load
  // batches (~16 b128 + 4 ds_load) stay in flight over every WMMA group.
  v16bf aF[3];
  v16bf bF[3][4];
#pragma unroll
  for (int p = 0; p < 2; ++p) {
    const int kk = p * 32;
    aF[p] = loadA(sArow + kk, khalf);
#pragma unroll
    for (int g = 0; g < 4; ++g) bF[p][g] = loadB(wrow[g] + kk, ksel);
  }
#pragma unroll
  for (int kk = 0; kk < K; kk += 32) {
    const int it  = kk >> 5;
    const int cur = it % 3;
    const int pf  = (it + 2) % 3;
    if (kk + 64 < K) {
      aF[pf] = loadA(sArow + kk + 64, khalf);
#pragma unroll
      for (int g = 0; g < 4; ++g) bF[pf][g] = loadB(wrow[g] + kk + 64, ksel);
    }
    __builtin_amdgcn_sched_barrier(0);   // keep prefetch loads above the WMMAs
    acc[0] = __builtin_amdgcn_wmma_f32_16x16x32_bf16(false, aF[cur], false, bF[cur][0], (short)0, acc[0], false, false);
    acc[1] = __builtin_amdgcn_wmma_f32_16x16x32_bf16(false, aF[cur], false, bF[cur][1], (short)0, acc[1], false, false);
    acc[2] = __builtin_amdgcn_wmma_f32_16x16x32_bf16(false, aF[cur], false, bF[cur][2], (short)0, acc[2], false, false);
    acc[3] = __builtin_amdgcn_wmma_f32_16x16x32_bf16(false, aF[cur], false, bF[cur][3], (short)0, acc[3], false, false);
    __builtin_amdgcn_sched_barrier(0);   // keep next iteration's loads below
  }

  // C/D layout (ISA 7.12.2): VGPR r -> M = r + 8*(lane>=16), N = lane%16.
  const int rbase = mt * 16 + (hi << 3);
#pragma unroll
  for (int r = 0; r < 8; ++r) {
    int row = rbase + r;
    int ci  = row * H_ + ncol;
    float iv = acc[0][r], fv = acc[1][r], gv = acc[2][r], ov = acc[3][r];
    float cn = sigmoidf_(fv) * c[ci] + sigmoidf_(iv) * tanhf(gv);
    c[ci] = cn;
    float h = sigmoidf_(ov) * tanhf(cn);
    u16 hb = f2bf(h);
    hA[(size_t)row * shA + ncol] = hb;
    if (hB) hB[(size_t)row * shB + ncol] = hb;
    if (hF) hF[ci] = h;
  }
}

// FC head: pv = relu(h1 @ W1^T + b1) @ W2^T + b2   (256x128x256 + 256x8x128)
__global__ __launch_bounds__(128) void fc_step(
    const float* __restrict__ h1,          // [B][H] fp32
    const float* __restrict__ W1, const float* __restrict__ b1,
    const float* __restrict__ W2, const float* __restrict__ b2,
    float* __restrict__ out,               // pre-offset to +t*NPV; [b*TT*NPV + v]
    u16*  __restrict__ pvOut)              // next act_dec0, cols 0..7
{
  __shared__ float sh[H_];
  __shared__ float sa[128];
  int b = blockIdx.x, tid = threadIdx.x;
  sh[tid]       = h1[b * H_ + tid];
  sh[tid + 128] = h1[b * H_ + tid + 128];
  __syncthreads();
  float acc = b1[tid];
  for (int k = 0; k < H_; ++k) acc += sh[k] * W1[tid * H_ + k];
  sa[tid] = fmaxf(acc, 0.0f);
  __syncthreads();
  if (tid < NPV_) {
    float o = b2[tid];
    for (int k = 0; k < 128; ++k) o += sa[k] * W2[tid * 128 + k];
    out[(size_t)b * (TT_ * NPV_) + tid] = o;
    pvOut[(size_t)b * 288 + tid] = f2bf(o);
  }
}

// ----------------------- prep / staging kernels ----------------------------
__global__ void fzero(float* p, int n) {
  int i = blockIdx.x * 256 + threadIdx.x; if (i < n) p[i] = 0.0f;
}
__global__ void uzero(u16* p, int n) {
  int i = blockIdx.x * 256 + threadIdx.x; if (i < n) p[i] = 0;
}
__global__ void cvt_copy(u16* dst, int dstStride, int dstOff,
                         const float* src, int srcStride, int srcOff,
                         int rows, int cols) {
  int i = blockIdx.x * 256 + threadIdx.x;
  if (i >= rows * cols) return;
  int r = i / cols, cc = i % cols;
  dst[(size_t)r * dstStride + dstOff + cc] =
      f2bf(src[(size_t)r * srcStride + srcOff + cc]);
}
__global__ void copy_us(u16* dst, int dstStride, int dstOff,
                        const u16* src, int srcStride, int srcOff,
                        int rows, int cols) {
  int i = blockIdx.x * 256 + threadIdx.x;
  if (i >= rows * cols) return;
  int r = i / cols, cc = i % cols;
  dst[(size_t)r * dstStride + dstOff + cc] =
      src[(size_t)r * srcStride + srcOff + cc];
}
// enc_in rows [t][b], 96 cols = [x_cv(64) | emb[scenario]](bf16)
__global__ void build_enc_in(u16* dst, const float* x_cv, const int* scen,
                             const float* emb) {
  size_t i = (size_t)blockIdx.x * 256 + threadIdx.x;
  if (i >= (size_t)TIN_ * B_ * 96) return;
  int k = (int)(i % 96);
  size_t rb = i / 96;
  int b = (int)(rb % B_), t = (int)(rb / B_);
  float v = (k < 64) ? x_cv[((size_t)b * TIN_ + t) * 64 + k]
                     : emb[scen[b] * 32 + (k - 64)];
  dst[i] = f2bf(v);
}
// dec_in rows [t][b], 64 cols = x_cv_target (bf16)
__global__ void build_dec_in(u16* dst, const float* x) {
  size_t i = (size_t)blockIdx.x * 256 + threadIdx.x;
  if (i >= (size_t)TT_ * B_ * 64) return;
  int k = (int)(i % 64);
  size_t rb = i / 64;
  int b = (int)(rb % B_), t = (int)(rb / B_);
  dst[i] = f2bf(x[((size_t)b * TT_ + t) * 64 + k]);
}

// ---------------------------------------------------------------------------
extern "C" void kernel_launch(void* const* d_in, const int* in_sizes, int n_in,
                              void* d_out, int out_size, void* d_ws, size_t ws_size,
                              hipStream_t stream) {
  (void)in_sizes; (void)n_in; (void)out_size; (void)ws_size;
  const float* x_cv  = (const float*)d_in[0];
  const float* x_tg  = (const float*)d_in[1];
  const float* pv0   = (const float*)d_in[2];
  const int*   scen  = (const int*)  d_in[3];
  const float* emb   = (const float*)d_in[4];
  const float* eWih0 = (const float*)d_in[5];
  const float* eWhh0 = (const float*)d_in[6];
  const float* eb0   = (const float*)d_in[7];
  const float* eWih1 = (const float*)d_in[8];
  const float* eWhh1 = (const float*)d_in[9];
  const float* eb1   = (const float*)d_in[10];
  const float* dWih0 = (const float*)d_in[11];
  const float* dWhh0 = (const float*)d_in[12];
  const float* db0   = (const float*)d_in[13];
  const float* dWih1 = (const float*)d_in[14];
  const float* dWhh1 = (const float*)d_in[15];
  const float* db1   = (const float*)d_in[16];
  const float* fW1   = (const float*)d_in[17];
  const float* fb1   = (const float*)d_in[18];
  const float* fW2   = (const float*)d_in[19];
  const float* fb2   = (const float*)d_in[20];
  float* out = (float*)d_out;

  // workspace carve-up (256B aligned)
  char* ws = (char*)d_ws;
  size_t off = 0;
  auto alloc = [&](size_t bytes) -> char* {
    char* p = ws + off;
    off += (bytes + 255) & ~(size_t)255;
    return p;
  };
  u16* enc_in = (u16*)alloc((size_t)TIN_ * B_ * 96 * 2);   // 25.2 MB
  u16* dec_in = (u16*)alloc((size_t)TT_  * B_ * 64 * 2);   //  8.4 MB
  u16* act[2] = { (u16*)alloc((size_t)B_ * 512 * 2),
                  (u16*)alloc((size_t)B_ * 512 * 2) };     // [h0|h1] ping-pong
  u16* d0[2]  = { (u16*)alloc((size_t)B_ * 288 * 2),
                  (u16*)alloc((size_t)B_ * 288 * 2) };     // [pv|h0|pad24]
  float* c0  = (float*)alloc((size_t)B_ * H_ * 4);
  float* c1  = (float*)alloc((size_t)B_ * H_ * 4);
  float* h1f = (float*)alloc((size_t)B_ * H_ * 4);
  u16* We0 = (u16*)alloc((size_t)G4H_ * 352 * 2);  // [enc_Wih0|enc_Whh0]
  u16* We1 = (u16*)alloc((size_t)G4H_ * 512 * 2);  // [enc_Wih1|enc_Whh1]
  u16* Wd0 = (u16*)alloc((size_t)G4H_ * 352 * 2);  // [dec_Wih0(72)|dec_Whh0|0pad]
  u16* Wd1 = (u16*)alloc((size_t)G4H_ * 512 * 2);  // [dec_Wih1|dec_Whh1]

  auto g1 = [](size_t n) { return dim3((unsigned)((n + 255) / 256)); };

  // ---- per-call prep (deterministic; cheap vs main loops) ----
  fzero<<<g1(B_*H_), 256, 0, stream>>>(c0, B_*H_);
  fzero<<<g1(B_*H_), 256, 0, stream>>>(c1, B_*H_);
  uzero<<<g1(B_*512), 256, 0, stream>>>(act[0], B_*512);
  uzero<<<g1(B_*288), 256, 0, stream>>>(d0[0], B_*288);
  uzero<<<g1(B_*288), 256, 0, stream>>>(d0[1], B_*288);
  uzero<<<g1((size_t)G4H_*352), 256, 0, stream>>>(Wd0, G4H_*352);  // zero K-pad
  cvt_copy<<<g1(B_*8), 256, 0, stream>>>(d0[0], 288, 0, pv0, 8, 0, B_, 8);
  cvt_copy<<<g1((size_t)G4H_*96),  256, 0, stream>>>(We0, 352, 0,  eWih0, 96,  0, G4H_, 96);
  cvt_copy<<<g1((size_t)G4H_*256), 256, 0, stream>>>(We0, 352, 96, eWhh0, 256, 0, G4H_, 256);
  cvt_copy<<<g1((size_t)G4H_*256), 256, 0, stream>>>(We1, 512, 0,   eWih1, 256, 0, G4H_, 256);
  cvt_copy<<<g1((size_t)G4H_*256), 256, 0, stream>>>(We1, 512, 256, eWhh1, 256, 0, G4H_, 256);
  cvt_copy<<<g1((size_t)G4H_*72),  256, 0, stream>>>(Wd0, 352, 0,  dWih0, 72,  0, G4H_, 72);
  cvt_copy<<<g1((size_t)G4H_*256), 256, 0, stream>>>(Wd0, 352, 72, dWhh0, 256, 0, G4H_, 256);
  cvt_copy<<<g1((size_t)G4H_*256), 256, 0, stream>>>(Wd1, 512, 0,   dWih1, 256, 0, G4H_, 256);
  cvt_copy<<<g1((size_t)G4H_*256), 256, 0, stream>>>(Wd1, 512, 256, dWhh1, 256, 0, G4H_, 256);
  build_enc_in<<<g1((size_t)TIN_*B_*96), 256, 0, stream>>>(enc_in, x_cv, scen, emb);
  build_dec_in<<<g1((size_t)TT_*B_*64),  256, 0, stream>>>(dec_in, x_tg);

  // ---- encoder: 512 steps, 2 layers, ping-pong h buffers ----
  int ci = 0;
  for (int t = 0; t < TIN_; ++t) {
    u16* cur = act[ci];
    u16* nxt = act[ci ^ 1];
    // layer0: A = [enc_in_t(96) | h0(256)], K=352
    lstm_step<352, 96><<<32, 256, 0, stream>>>(
        enc_in + (size_t)t * B_ * 96, 96, cur, 512,
        We0, eb0, c0, nxt, 512, nullptr, 0, nullptr);
    // layer1: A = [h0_new(256) from nxt | h1_prev(256) from cur], K=512
    lstm_step<512, 256><<<32, 256, 0, stream>>>(
        nxt, 512, cur + 256, 512,
        We1, eb1, c1, nxt + 256, 512, nullptr, 0, nullptr);
    ci ^= 1;
  }

  // bridge: decoder layer0 recurrent buffer gets encoder-final h0
  copy_us<<<g1((size_t)B_*H_), 256, 0, stream>>>(d0[0], 288, 8,
                                                 act[ci], 512, 0, B_, H_);

  // ---- decoder: 256 steps, 2 layers + FC head ----
  int di = 0;
  for (int t = 0; t < TT_; ++t) {
    u16* cur  = act[ci];
    u16* nxt  = act[ci ^ 1];
    u16* dcur = d0[di];
    u16* dnxt = d0[di ^ 1];
    // layer0: A = [x_t(64) | pv(8)+h0(256)+pad(24)], K=352
    lstm_step<352, 64><<<32, 256, 0, stream>>>(
        dec_in + (size_t)t * B_ * 64, 64, dcur, 288,
        Wd0, db0, c0, dnxt + 8, 288, nxt, 512, nullptr);
    // layer1: A = [h0_new | h1_prev], K=512; also emit fp32 h1 for FC
    lstm_step<512, 256><<<32, 256, 0, stream>>>(
        nxt, 512, cur + 256, 512,
        Wd1, db1, c1, nxt + 256, 512, nullptr, 0, h1f);
    // FC head -> d_out[:, t, :] and pv feedback into dnxt cols 0..7
    fc_step<<<B_, 128, 0, stream>>>(h1f, fW1, fb1, fW2, fb2,
                                    out + (size_t)t * NPV_, dnxt);
    ci ^= 1;
    di ^= 1;
  }
}
